// MixAttention_4234837754377
// MI455X (gfx1250) — compile-verified, matrix operand order checked by reading, not run on hardware
//
#include <hip/hip_runtime.h>
#include <math.h>

// ---------------------------------------------------------------------------
// MixAttention block for MI455X (gfx1250), wave32, WMMA f32_16x16x32_f16.
// Workspace requirement: 4 * 128MiB big buffers + ~2MiB small = ~539 MiB.
// ---------------------------------------------------------------------------

typedef _Float16 f16;
typedef __attribute__((ext_vector_type(16))) _Float16 v16h;
typedef __attribute__((ext_vector_type(8)))  float    v8f;

#define DIM   128
#define HEADS 4
#define HD    32
#define WS_   8
#define NTOK  64
#define NWIN  4096
#define IMG   256
#define NB    4
#define EPS_  1e-5f

// ---- WMMA fragment helpers -------------------------------------------------
// A-fragment (16x32 f16) from a row-major matrix: element (m,k) at p[m*lda+k].
// ISA layout: lanes 0-15 hold row m=l, K 0..7 & 16..23; lanes 16-31 hold
// row m=l-16, K 8..15 & 24..31.
__device__ __forceinline__ v16h frag_row(const f16* p, int lda, int r0, int k0) {
  const int l = threadIdx.x & 31;
  const f16* q = p + (size_t)(r0 + (l & 15)) * lda + k0 + ((l >> 4) << 3);
  v16h v;
#pragma unroll
  for (int i = 0; i < 8; ++i) { v[i] = q[i]; v[i + 8] = q[i + 16]; }
  return v;
}

// B-fragment from a K-major matrix: element (k,n) at p[k*ldb+n] (i.e. B as
// stored K x N; column n strided).
__device__ __forceinline__ v16h frag_col(const f16* p, int ldb, int n0, int k0) {
  const int l = threadIdx.x & 31;
  const f16* q = p + (size_t)(k0 + ((l >> 4) << 3)) * ldb + n0 + (l & 15);
  v16h v;
#pragma unroll
  for (int i = 0; i < 8; ++i) { v[i] = q[(size_t)i * ldb]; v[i + 8] = q[(size_t)(i + 16) * ldb]; }
  return v;
}

__device__ __forceinline__ v8f wmma16(v16h a, v16h b, v8f c) {
  // D = A x B + C  -> v_wmma_f32_16x16x32_f16
  return __builtin_amdgcn_wmma_f32_16x16x32_f16(false, a, false, b, (short)0, c,
                                                false, false);
}

// C/D tile (16x16 f32): VGPR r, lanes 0-15 -> (m0+r, n0+lane); lanes 16-31 ->
// (m0+8+r, n0+lane-16).
__device__ __forceinline__ void frag_store_f32(float* p, int ldd, int m0, int n0, v8f c) {
  const int l = threadIdx.x & 31;
  float* q = p + (size_t)(m0 + ((l >> 4) << 3)) * ldd + n0 + (l & 15);
#pragma unroll
  for (int r = 0; r < 8; ++r) q[(size_t)r * ldd] = c[r];
}

__device__ __forceinline__ float gelu_ex(float v) {
  return 0.5f * v * (1.0f + erff(v * 0.70710678118654752f));
}

// ---- weight f32 -> f16 conversion ------------------------------------------
__global__ void k_f32_to_f16(const float* __restrict__ src, f16* __restrict__ dst, int n) {
  int i = blockIdx.x * blockDim.x + threadIdx.x;
  if (i < n) dst[i] = (f16)src[i];
}

// ---- fused 3-way projection + LayerNorm + window-reverse --------------------
__global__ void __launch_bounds__(256)
k_proj3_ln(const float* __restrict__ x,
           const f16* __restrict__ wPa, const f16* __restrict__ wPc1, const f16* __restrict__ wPc2,
           const float* __restrict__ bPa, const float* __restrict__ bPc1, const float* __restrict__ bPc2,
           const float* __restrict__ gLa, const float* __restrict__ bLa,
           const float* __restrict__ gL1, const float* __restrict__ bL1,
           const float* __restrict__ gL2, const float* __restrict__ bL2,
           float* __restrict__ outAtt, float* __restrict__ outC1, float* __restrict__ outC2) {
  __shared__ __align__(16) unsigned char smem[17408 + 32768];
  f16*   sX = (f16*)smem;                // 64 x 136 (padded)
  f16*   sW = (f16*)(smem + 17408);      // 128 x 128
  float* sO = (float*)(smem + 17408);    // 64 x 128 (overlays sW, lifetime disjoint)
  const int tid = threadIdx.x, wv = tid >> 5;
  const int win = blockIdx.x;
  const float* xw = x + (size_t)win * NTOK * DIM;
  for (int i = tid; i < NTOK * DIM; i += 256)
    sX[(i >> 7) * 136 + (i & 127)] = (f16)xw[i];
  __syncthreads();

  for (int sel = 0; sel < 3; ++sel) {
    const f16* Wg = (sel == 0) ? wPa : (sel == 1 ? wPc1 : wPc2);
    for (int i = tid; i < DIM * DIM; i += 256) sW[i] = Wg[i];
    __syncthreads();

    v8f acc[4];
    const v8f vz = {0.f,0.f,0.f,0.f,0.f,0.f,0.f,0.f};
#pragma unroll
    for (int m = 0; m < 4; ++m) acc[m] = vz;
    for (int k = 0; k < DIM; k += 32) {
      v16h bfr = frag_row(sW, DIM, wv * 16, k);
#pragma unroll
      for (int m = 0; m < 4; ++m)
        acc[m] = wmma16(frag_row(sX, 136, m * 16, k), bfr, acc[m]);
    }
    __syncthreads();   // sW dead -> sO may overlay
#pragma unroll
    for (int m = 0; m < 4; ++m) frag_store_f32(sO, DIM, m * 16, wv * 16, acc[m]);
    __syncthreads();

    if (tid < NTOK) {
      const float* bias = (sel == 0) ? bPa : (sel == 1 ? bPc1 : bPc2);
      const float* g    = (sel == 0) ? gLa : (sel == 1 ? gL1 : gL2);
      const float* bb   = (sel == 0) ? bLa : (sel == 1 ? bL1 : bL2);
      float mu = 0.f;
      for (int c = 0; c < DIM; ++c) mu += sO[tid * DIM + c] + bias[c];
      mu *= (1.0f / DIM);
      float var = 0.f;
      for (int c = 0; c < DIM; ++c) {
        float d = sO[tid * DIM + c] + bias[c] - mu; var += d * d;
      }
      var *= (1.0f / DIM);
      float rs = rsqrtf(var + EPS_);
      if (sel == 0) {
        float* o = outAtt + (size_t)win * NTOK * DIM + (size_t)tid * DIM;
        for (int c = 0; c < DIM; ++c)
          o[c] = (sO[tid * DIM + c] + bias[c] - mu) * rs * g[c] + bb[c];
      } else {
        float* dst = (sel == 1) ? outC1 : outC2;
        int b = win >> 10, hb = (win >> 5) & 31, wb = win & 31;
        int ih = hb * WS_ + (tid >> 3), iw = wb * WS_ + (tid & 7);
        for (int c = 0; c < DIM; ++c)
          dst[(((size_t)b * DIM + c) * IMG + ih) * IMG + iw] =
              (sO[tid * DIM + c] + bias[c] - mu) * rs * g[c] + bb[c];
      }
    }
    __syncthreads();
  }
}

// ---- depthwise conv (SAME) + BN + GELU -------------------------------------
template <int KS>
__global__ void __launch_bounds__(256)
k_dw_bn_gelu(const float* __restrict__ src, const float* __restrict__ w,
             const float* __restrict__ cb, const float* __restrict__ g,
             const float* __restrict__ bb, float* __restrict__ dst) {
  const int h  = blockIdx.x % IMG;
  const int bc = blockIdx.x / IMG;         // b*DIM + c
  const int c  = bc % DIM;
  const int xx0 = threadIdx.x;
  const float* plane = src + (size_t)bc * IMG * IMG;
  const int P = KS / 2;
  float acc = 0.f;
#pragma unroll
  for (int ky = 0; ky < KS; ++ky) {
    int yy = h + ky - P;
    if (yy < 0 || yy >= IMG) continue;
#pragma unroll
    for (int kx = 0; kx < KS; ++kx) {
      int xc = xx0 + kx - P;
      if (xc < 0 || xc >= IMG) continue;
      acc += plane[yy * IMG + xc] * w[c * KS * KS + ky * KS + kx];
    }
  }
  float s = g[c] * rsqrtf(1.0f + EPS_);
  float v = (acc + cb[c]) * s + bb[c];
  dst[(size_t)bc * IMG * IMG + (size_t)h * IMG + xx0] = gelu_ex(v);
}

// ---- 1x1 conv (128 -> 64) + BN over 64-pixel tiles via WMMA ----------------
__global__ void __launch_bounds__(128)
k_conv1x1_bn(const float* __restrict__ src, const f16* __restrict__ w,
             const float* __restrict__ cb, const float* __restrict__ g,
             const float* __restrict__ bb, float* __restrict__ dst, int obase) {
  __shared__ f16 sA[NTOK][DIM];
  __shared__ f16 sW[64][DIM];
  __shared__ float sO[NTOK][64];
  const int tid = threadIdx.x, wv = tid >> 5;
  const int pix0 = blockIdx.x * 64;        // flat over NB*IMG*IMG, W-inner
  const int w0 = pix0 & (IMG - 1);
  const int bh = pix0 / IMG;
  const int h = bh % IMG, b = bh / IMG;
  for (int i = tid; i < 64 * DIM; i += 128) {
    int c = i >> 6, m = i & 63;
    sA[m][c] = (f16)src[(((size_t)b * DIM + c) * IMG + h) * IMG + w0 + m];
  }
  for (int i = tid; i < 64 * DIM; i += 128) sW[i >> 7][i & 127] = w[i];
  __syncthreads();
  const v8f vz = {0.f,0.f,0.f,0.f,0.f,0.f,0.f,0.f};
  v8f acc[4];
#pragma unroll
  for (int m = 0; m < 4; ++m) acc[m] = vz;
  for (int k = 0; k < DIM; k += 32) {
    v16h bfr = frag_row(&sW[0][0], DIM, wv * 16, k);
#pragma unroll
    for (int m = 0; m < 4; ++m)
      acc[m] = wmma16(frag_row(&sA[0][0], DIM, m * 16, k), bfr, acc[m]);
  }
#pragma unroll
  for (int m = 0; m < 4; ++m) frag_store_f32(&sO[0][0], 64, m * 16, wv * 16, acc[m]);
  __syncthreads();
  for (int i = tid; i < 64 * 64; i += 128) {
    int n = i >> 6, m = i & 63;
    float sc = g[n] * rsqrtf(1.0f + EPS_);
    float v = (sO[m][n] + cb[n]) * sc + bb[n];
    dst[(((size_t)b * DIM + obase + n) * IMG + h) * IMG + w0 + m] = v;
  }
}

// ---- global average pool per (b,c) -----------------------------------------
__global__ void __launch_bounds__(256)
k_pool(const float* __restrict__ xcnn, float* __restrict__ pooled) {
  __shared__ float red[256];
  const int bc = blockIdx.x;
  const float* p = xcnn + (size_t)bc * IMG * IMG;
  float s = 0.f;
  for (int i = threadIdx.x; i < IMG * IMG; i += 256) s += p[i];
  red[threadIdx.x] = s;
  __syncthreads();
  for (int st = 128; st > 0; st >>= 1) {
    if (threadIdx.x < st) red[threadIdx.x] += red[threadIdx.x + st];
    __syncthreads();
  }
  if (threadIdx.x == 0) pooled[bc] = red[0] * (1.0f / (IMG * IMG));
}

// ---- channel-interaction MLP -> sigmoid gate (4,128) -----------------------
__global__ void __launch_bounds__(128)
k_ci_gate(const float* __restrict__ pooled,
          const float* __restrict__ w1, const float* __restrict__ b1,
          const float* __restrict__ g1, const float* __restrict__ bb1,
          const float* __restrict__ w2, const float* __restrict__ b2,
          float* __restrict__ gate) {
  __shared__ float hid[64];
  for (int b = 0; b < NB; ++b) {
    if (threadIdx.x < 64) {
      float a = 0.f;
      for (int c = 0; c < DIM; ++c) a += pooled[b * DIM + c] * w1[threadIdx.x * DIM + c];
      float sc = g1[threadIdx.x] * rsqrtf(1.0f + EPS_);
      float v = (a + b1[threadIdx.x]) * sc + bb1[threadIdx.x];
      hid[threadIdx.x] = gelu_ex(v);
    }
    __syncthreads();
    {
      float a = 0.f;
      for (int h = 0; h < 64; ++h) a += hid[h] * w2[threadIdx.x * 64 + h];
      a += b2[0];
      gate[b * DIM + threadIdx.x] = 1.0f / (1.0f + __expf(-a));
    }
    __syncthreads();
  }
}

// ---- window attention: qkv GEMM + scores + softmax + P*V, in-place ---------
__global__ void __launch_bounds__(256)
k_attn(float* __restrict__ xatt, const f16* __restrict__ wqkv,
       const float* __restrict__ bqkv, const float* __restrict__ rpb,
       const float* __restrict__ gate) {
  __shared__ f16 sX[NTOK][DIM];
  __shared__ f16 sQ[NTOK][HD];
  __shared__ f16 sK[NTOK][HD];
  __shared__ f16 sV[NTOK][HD];
  __shared__ float sS[NTOK][NTOK];
  __shared__ f16 sP[NTOK][NTOK];
  const int tid = threadIdx.x, wv = tid >> 5, ln = tid & 31;
  const int win = blockIdx.x, b = win >> 10;
  float* xw = xatt + (size_t)win * NTOK * DIM;
  for (int i = tid; i < NTOK * DIM; i += 256) sX[i >> 7][i & 127] = (f16)xw[i];
  __syncthreads();
  const v8f vz = {0.f,0.f,0.f,0.f,0.f,0.f,0.f,0.f};

  for (int hd = 0; hd < HEADS; ++hd) {
    // q,k,v projection for this head: 3 x (4 Mtiles x 2 Ntiles) = 24 tiles
    for (int t = wv * 3; t < wv * 3 + 3; ++t) {
      int sel = t >> 3;                 // 0=q 1=k 2=v
      int rem = t & 7;
      int mt = rem >> 1, nt = rem & 1;
      int nrow = sel * DIM + hd * HD + nt * 16;   // weight row (out channel)
      v8f acc = vz;
      for (int k = 0; k < DIM; k += 32)
        acc = wmma16(frag_row(&sX[0][0], DIM, mt * 16, k),
                     frag_row(wqkv, DIM, nrow, k), acc);
      int ncol = nt * 16 + (ln & 15);
      int m0 = mt * 16 + ((ln >> 4) << 3);
      float bias = bqkv[sel * DIM + hd * HD + ncol];
      f16* dstp = (sel == 0) ? &sQ[0][0] : (sel == 1 ? &sK[0][0] : &sV[0][0]);
      float mul = 1.0f;
      if (sel == 0) mul = 0.17677669529663687f;          // HD^-0.5
      if (sel == 2) mul = gate[b * DIM + hd * HD + ncol]; // sigmoid channel gate
#pragma unroll
      for (int r = 0; r < 8; ++r)
        dstp[(m0 + r) * HD + ncol] = (f16)((acc[r] + bias) * mul);
    }
    __syncthreads();

    // scores = q k^T + rel-pos bias  (K = HD = 32 -> one WMMA per tile)
    for (int t = wv * 2; t < wv * 2 + 2; ++t) {
      int mt = t >> 2, nt = t & 3;
      v8f acc = wmma16(frag_row(&sQ[0][0], HD, mt * 16, 0),
                       frag_row(&sK[0][0], HD, nt * 16, 0), vz);
      int n = nt * 16 + (ln & 15);
      int m0 = mt * 16 + ((ln >> 4) << 3);
      int nih = n >> 3, niw = n & 7;
#pragma unroll
      for (int r = 0; r < 8; ++r) {
        int m = m0 + r;
        int idx = ((m >> 3) - nih + 7) * 15 + ((m & 7) - niw + 7);
        sS[m][n] = acc[r] + rpb[idx * HEADS + hd];
      }
    }
    __syncthreads();

    // row softmax
    if (tid < NTOK) {
      float mx = -1e30f;
      for (int n = 0; n < NTOK; ++n) mx = fmaxf(mx, sS[tid][n]);
      float sm = 0.f;
      for (int n = 0; n < NTOK; ++n) sm += __expf(sS[tid][n] - mx);
      float inv = 1.0f / sm;
      for (int n = 0; n < NTOK; ++n)
        sP[tid][n] = (f16)(__expf(sS[tid][n] - mx) * inv);
    }
    __syncthreads();

    // out = P (64x64) @ V (64x32): 8 tiles, 1 per wave
    {
      int mt = wv >> 1, nt = wv & 1;
      v8f acc = vz;
      for (int k = 0; k < NTOK; k += 32)
        acc = wmma16(frag_row(&sP[0][0], NTOK, mt * 16, k),
                     frag_col(&sV[0][0], HD, nt * 16, k), acc);
      int n = nt * 16 + (ln & 15);
      int m0 = mt * 16 + ((ln >> 4) << 3);
#pragma unroll
      for (int r = 0; r < 8; ++r)
        xw[(size_t)(m0 + r) * DIM + hd * HD + n] = acc[r];
    }
    __syncthreads();
  }
}

// ---- spatial-interaction gate: conv1x1(128->64)+BN+GELU then dot -> sigmoid -
__global__ void __launch_bounds__(128)
k_si(const float* __restrict__ att, const f16* __restrict__ w1,
     const float* __restrict__ b1, const float* __restrict__ g1,
     const float* __restrict__ bb1, const float* __restrict__ w2,
     const float* __restrict__ b2, float* __restrict__ sig) {
  __shared__ f16 sA[NTOK][DIM];
  __shared__ f16 sW[64][DIM];
  __shared__ float sO[NTOK][64];
  __shared__ f16 sH[NTOK][64];
  const int tid = threadIdx.x, wv = tid >> 5;
  const int win = blockIdx.x;
  const float* xw = att + (size_t)win * NTOK * DIM;
  for (int i = tid; i < NTOK * DIM; i += 128) sA[i >> 7][i & 127] = (f16)xw[i];
  for (int i = tid; i < 64 * DIM; i += 128) sW[i >> 7][i & 127] = w1[i];
  __syncthreads();
  const v8f vz = {0.f,0.f,0.f,0.f,0.f,0.f,0.f,0.f};
  v8f acc[4];
#pragma unroll
  for (int m = 0; m < 4; ++m) acc[m] = vz;
  for (int k = 0; k < DIM; k += 32) {
    v16h bfr = frag_row(&sW[0][0], DIM, wv * 16, k);
#pragma unroll
    for (int m = 0; m < 4; ++m)
      acc[m] = wmma16(frag_row(&sA[0][0], DIM, m * 16, k), bfr, acc[m]);
  }
#pragma unroll
  for (int m = 0; m < 4; ++m) frag_store_f32(&sO[0][0], 64, m * 16, wv * 16, acc[m]);
  __syncthreads();
  for (int i = tid; i < NTOK * 64; i += 128) {
    int m = i >> 6, n = i & 63;
    float sc = g1[n] * rsqrtf(1.0f + EPS_);
    float v = (sO[m][n] + b1[n]) * sc + bb1[n];
    sH[m][n] = (f16)gelu_ex(v);
  }
  __syncthreads();
  if (tid < NTOK) {
    float a = 0.f;
    for (int h = 0; h < 64; ++h) a += (float)sH[tid][h] * w2[h];
    a += b2[0];
    sig[win * NTOK + tid] = 1.0f / (1.0f + __expf(-a));
  }
}

// ---- final: gate * x_cnn, dual LN, concat GEMM (K=256), proj GEMM ----------
__global__ void __launch_bounds__(256)
k_final(const float* __restrict__ xcnn, const float* __restrict__ att,
        const float* __restrict__ sig,
        const float* __restrict__ gcv, const float* __restrict__ bcv,
        const float* __restrict__ gat, const float* __restrict__ bat,
        const f16* __restrict__ wcat, const float* __restrict__ bcat,
        const f16* __restrict__ wproj, const float* __restrict__ bproj,
        float* __restrict__ out) {
  __shared__ f16 sCat[NTOK][2 * DIM];
  __shared__ f16 sMid[NTOK][DIM];
  const int tid = threadIdx.x, wv = tid >> 5, ln = tid & 31;
  const int win = blockIdx.x;
  const int b = win >> 10, hb = (win >> 5) & 31, wb = win & 31;

  if (tid < 128) {
    int t = tid & 63, half = tid >> 6;
    if (half == 0) {  // gated x_cnn gather (win_part fused) + LN(lcv)
      int ih = hb * WS_ + (t >> 3), iw = wb * WS_ + (t & 7);
      float sg = sig[win * NTOK + t];
      size_t base = ((size_t)b * DIM) * IMG * IMG + (size_t)ih * IMG + iw;
      float mu = 0.f;
      for (int c = 0; c < DIM; ++c) mu += xcnn[base + (size_t)c * IMG * IMG] * sg;
      mu *= (1.0f / DIM);
      float var = 0.f;
      for (int c = 0; c < DIM; ++c) {
        float d = xcnn[base + (size_t)c * IMG * IMG] * sg - mu; var += d * d;
      }
      var *= (1.0f / DIM);
      float rs = rsqrtf(var + EPS_);
      for (int c = 0; c < DIM; ++c)
        sCat[t][c] = (f16)((xcnn[base + (size_t)c * IMG * IMG] * sg - mu) * rs * gcv[c] + bcv[c]);
    } else {          // LN(lat) of attention output
      const float* row = att + (size_t)win * NTOK * DIM + (size_t)t * DIM;
      float mu = 0.f;
      for (int c = 0; c < DIM; ++c) mu += row[c];
      mu *= (1.0f / DIM);
      float var = 0.f;
      for (int c = 0; c < DIM; ++c) { float d = row[c] - mu; var += d * d; }
      var *= (1.0f / DIM);
      float rs = rsqrtf(var + EPS_);
      for (int c = 0; c < DIM; ++c)
        sCat[t][DIM + c] = (f16)((row[c] - mu) * rs * gat[c] + bat[c]);
    }
  }
  __syncthreads();
  const v8f vz = {0.f,0.f,0.f,0.f,0.f,0.f,0.f,0.f};

  {  // GEMM1: (64x256) @ (256x128)^T -> sMid (+bias, f16)
    v8f acc[4];
#pragma unroll
    for (int m = 0; m < 4; ++m) acc[m] = vz;
    for (int k = 0; k < 2 * DIM; k += 32) {
      v16h bfr = frag_row(wcat, 2 * DIM, wv * 16, k);
#pragma unroll
      for (int m = 0; m < 4; ++m)
        acc[m] = wmma16(frag_row(&sCat[0][0], 2 * DIM, m * 16, k), bfr, acc[m]);
    }
    int n = wv * 16 + (ln & 15);
    float bias = bcat[n];
#pragma unroll
    for (int m = 0; m < 4; ++m) {
      int m0 = m * 16 + ((ln >> 4) << 3);
#pragma unroll
      for (int r = 0; r < 8; ++r) sMid[m0 + r][n] = (f16)(acc[m][r] + bias);
    }
  }
  __syncthreads();
  {  // GEMM2: (64x128) @ (128x128)^T -> out (+bias, f32)
    v8f acc[4];
#pragma unroll
    for (int m = 0; m < 4; ++m) acc[m] = vz;
    for (int k = 0; k < DIM; k += 32) {
      v16h bfr = frag_row(wproj, DIM, wv * 16, k);
#pragma unroll
      for (int m = 0; m < 4; ++m)
        acc[m] = wmma16(frag_row(&sMid[0][0], DIM, m * 16, k), bfr, acc[m]);
    }
    int n = wv * 16 + (ln & 15);
    float bias = bproj[n];
    float* o = out + (size_t)win * NTOK * DIM;
#pragma unroll
    for (int m = 0; m < 4; ++m) {
      int m0 = m * 16 + ((ln >> 4) << 3);
#pragma unroll
      for (int r = 0; r < 8; ++r) o[(size_t)(m0 + r) * DIM + n] = acc[m][r] + bias;
    }
  }
}

// ---------------------------------------------------------------------------
extern "C" void kernel_launch(void* const* d_in, const int* in_sizes, int n_in,
                              void* d_out, int out_size, void* d_ws, size_t ws_size,
                              hipStream_t stream) {
  (void)in_sizes; (void)n_in; (void)out_size; (void)ws_size;
  // dict order: x, H, W, then params in insertion order
  const float* x      = (const float*)d_in[0];
  const float* rpb    = (const float*)d_in[3];
  const float* pa_w   = (const float*)d_in[4];
  const float* pa_b   = (const float*)d_in[5];
  const float* la_g   = (const float*)d_in[6];
  const float* la_b   = (const float*)d_in[7];
  const float* pc1_w  = (const float*)d_in[8];
  const float* pc1_b  = (const float*)d_in[9];
  const float* lc1_g  = (const float*)d_in[10];
  const float* lc1_b  = (const float*)d_in[11];
  const float* pc2_w  = (const float*)d_in[12];
  const float* pc2_b  = (const float*)d_in[13];
  const float* lc2_g  = (const float*)d_in[14];
  const float* lc2_b  = (const float*)d_in[15];
  const float* dw3_w  = (const float*)d_in[16];
  const float* dw3_b  = (const float*)d_in[17];
  const float* bn3_g  = (const float*)d_in[18];
  const float* bn3_b  = (const float*)d_in[19];
  const float* dw5_w  = (const float*)d_in[20];
  const float* dw5_b  = (const float*)d_in[21];
  const float* bn5_g  = (const float*)d_in[22];
  const float* bn5_b  = (const float*)d_in[23];
  const float* ci1_w  = (const float*)d_in[24];
  const float* ci1_b  = (const float*)d_in[25];
  const float* cibn_g = (const float*)d_in[26];
  const float* cibn_b = (const float*)d_in[27];
  const float* ci2_w  = (const float*)d_in[28];
  const float* ci2_b  = (const float*)d_in[29];
  const float* p1_w   = (const float*)d_in[30];
  const float* p1_b   = (const float*)d_in[31];
  const float* cn1_g  = (const float*)d_in[32];
  const float* cn1_b  = (const float*)d_in[33];
  const float* p2_w   = (const float*)d_in[34];
  const float* p2_b   = (const float*)d_in[35];
  const float* cn2_g  = (const float*)d_in[36];
  const float* cn2_b  = (const float*)d_in[37];
  const float* qkv_w  = (const float*)d_in[38];
  const float* qkv_b  = (const float*)d_in[39];
  const float* si1_w  = (const float*)d_in[40];
  const float* si1_b  = (const float*)d_in[41];
  const float* sibn_g = (const float*)d_in[42];
  const float* sibn_b = (const float*)d_in[43];
  const float* si2_w  = (const float*)d_in[44];
  const float* si2_b  = (const float*)d_in[45];
  const float* lat_g  = (const float*)d_in[46];
  const float* lat_b  = (const float*)d_in[47];
  const float* lcv_g  = (const float*)d_in[48];
  const float* lcv_b  = (const float*)d_in[49];
  const float* cat_w  = (const float*)d_in[50];
  const float* cat_b  = (const float*)d_in[51];
  const float* proj_w = (const float*)d_in[52];
  const float* proj_b = (const float*)d_in[53];

  // workspace layout
  const size_t BIG = (size_t)NWIN * NTOK * DIM * sizeof(float);  // 128 MiB
  char* ws = (char*)d_ws;
  float* bufA = (float*)(ws + 0 * BIG);  // x_atten, then attention output (in-place)
  float* bufB = (float*)(ws + 1 * BIG);  // x_cnn1_pre, then concatenated x_cnn
  float* bufC = (float*)(ws + 2 * BIG);  // x_cnn2_pre
  float* bufD = (float*)(ws + 3 * BIG);  // depthwise-conv temp (reused)
  char* sm = ws + 4 * BIG;
  f16* wPa   = (f16*)sm; sm += (size_t)DIM * DIM * 2;
  f16* wPc1  = (f16*)sm; sm += (size_t)DIM * DIM * 2;
  f16* wPc2  = (f16*)sm; sm += (size_t)DIM * DIM * 2;
  f16* wQkv  = (f16*)sm; sm += (size_t)3 * DIM * DIM * 2;
  f16* wP1   = (f16*)sm; sm += (size_t)64 * DIM * 2;
  f16* wP2   = (f16*)sm; sm += (size_t)64 * DIM * 2;
  f16* wSi1  = (f16*)sm; sm += (size_t)64 * DIM * 2;
  f16* wCat  = (f16*)sm; sm += (size_t)DIM * 2 * DIM * 2;
  f16* wProj = (f16*)sm; sm += (size_t)DIM * DIM * 2;
  float* pooled = (float*)sm; sm += NB * DIM * 4;
  float* gate   = (float*)sm; sm += NB * DIM * 4;
  float* sig    = (float*)sm; sm += (size_t)NWIN * NTOK * 4;

  auto cvt = [&](const float* s, f16* d, int n) {
    k_f32_to_f16<<<(n + 255) / 256, 256, 0, stream>>>(s, d, n);
  };
  cvt(pa_w,  wPa,   DIM * DIM);
  cvt(pc1_w, wPc1,  DIM * DIM);
  cvt(pc2_w, wPc2,  DIM * DIM);
  cvt(qkv_w, wQkv,  3 * DIM * DIM);
  cvt(p1_w,  wP1,   64 * DIM);
  cvt(p2_w,  wP2,   64 * DIM);
  cvt(si1_w, wSi1,  64 * DIM);
  cvt(cat_w, wCat,  DIM * 2 * DIM);
  cvt(proj_w, wProj, DIM * DIM);

  // 1) three projections + LN (+ fused window-reverse for CNN branches)
  k_proj3_ln<<<NWIN, 256, 0, stream>>>(x, wPa, wPc1, wPc2, pa_b, pc1_b, pc2_b,
                                       la_g, la_b, lc1_g, lc1_b, lc2_g, lc2_b,
                                       bufA, bufB, bufC);
  // 2) CNN branch 1: dw3 + BN + GELU, then 1x1 conv + BN -> channels [0,64)
  k_dw_bn_gelu<3><<<NB * DIM * IMG, 256, 0, stream>>>(bufB, dw3_w, dw3_b, bn3_g, bn3_b, bufD);
  k_conv1x1_bn<<<NB * IMG * IMG / 64, 128, 0, stream>>>(bufD, wP1, p1_b, cn1_g, cn1_b, bufB, 0);
  // 3) CNN branch 2: dw5 + BN + GELU, then 1x1 conv + BN -> channels [64,128)
  k_dw_bn_gelu<5><<<NB * DIM * IMG, 256, 0, stream>>>(bufC, dw5_w, dw5_b, bn5_g, bn5_b, bufD);
  k_conv1x1_bn<<<NB * IMG * IMG / 64, 128, 0, stream>>>(bufD, wP2, p2_b, cn2_g, cn2_b, bufB, 64);
  // 4) channel gate
  k_pool<<<NB * DIM, 256, 0, stream>>>(bufB, pooled);
  k_ci_gate<<<1, 128, 0, stream>>>(pooled, ci1_w, ci1_b, cibn_g, cibn_b, ci2_w, ci2_b, gate);
  // 5) window attention (in-place over bufA)
  k_attn<<<NWIN, 256, 0, stream>>>(bufA, wQkv, qkv_b, rpb, gate);
  // 6) spatial gate
  k_si<<<NWIN, 128, 0, stream>>>(bufA, wSi1, si1_b, sibn_g, sibn_b, si2_w, si2_b, sig);
  // 7) gated concat + dual LN + cat GEMM + proj GEMM
  k_final<<<NWIN, 256, 0, stream>>>(bufB, bufA, sig, lcv_g, lcv_b, lat_g, lat_b,
                                    wCat, cat_b, wProj, proj_b, (float*)d_out);
}
